// Attention_76716705841940
// MI455X (gfx1250) — matrix operand order repaired and checked by reading, not compiled, not run on hardware
//
#include <hip/hip_runtime.h>
#include <hip/hip_bf16.h>
#include <math.h>

// ---------------------------------------------------------------------------
// Single-head causal attention (nanoGPT Head) for MI455X / gfx1250.
//   x:[8,2048,1024] f32, Wq/Wk/Wv:[1024,128] f32 -> out:[8,2048,128] f32
// bf16 WMMA (v_wmma_f32_16x16x32_bf16, f32 accum) + flash-attention tiling.
// x read once per projection (fused QKV kernel); Q,K row-major bf16, V transposed bf16 so every
// WMMA fragment is a contiguous 16/32-byte per-lane load.  Softmax stats
// (m, l, alpha) are register-resident with half-wave shuffle reductions; LDS
// is used only for the C-layout -> A-layout transpose of P (stored bf16).
// ---------------------------------------------------------------------------

typedef __bf16 bf16;
typedef __attribute__((ext_vector_type(16))) __bf16 v16bf;
typedef __attribute__((ext_vector_type(8)))  __bf16 v8bf;
typedef __attribute__((ext_vector_type(8)))  float  v8f;

#define N_EMBED 1024
#define HEAD    128
#define SEQ     2048
#define NBATCH  8
#define BT      (NBATCH * SEQ)   // 16384 rows

static __device__ __forceinline__ v8f wmma_bf16(v16bf a, v16bf b, v8f c) {
  // (neg_a, A, neg_b, B, c_mod, C, reuse_a, reuse_b)
  return __builtin_amdgcn_wmma_f32_16x16x32_bf16(false, a, false, b,
                                                 (short)0, c, false, false);
}

static __device__ __forceinline__ v16bf concat8(v8bf lo, v8bf hi) {
  return __builtin_shufflevector(lo, hi, 0, 1, 2, 3, 4, 5, 6, 7,
                                 8, 9, 10, 11, 12, 13, 14, 15);
}

// A-matrix 16x32 bf16 fragment from bf16 row-major source (row stride ld).
// ISA layout: lane<16 -> row=lane, K={0..7,16..23}; lane>=16 -> row=lane-16,
// K={8..15,24..31}.  Two 16-byte loads per lane.
static __device__ __forceinline__ v16bf load_a_bf16(const bf16* base, int ld,
                                                    int lane) {
  const int row = lane & 15;
  const int kb  = (lane < 16) ? 0 : 8;
  const bf16* p0 = base + row * ld + kb;
  v8bf lo = *reinterpret_cast<const v8bf*>(p0);
  v8bf hi = *reinterpret_cast<const v8bf*>(p0 + 16);
  return concat8(lo, hi);
}

// A-matrix fragment from an f32 row-major source, cvt f32->bf16 on the fly.
static __device__ __forceinline__ v16bf load_a_f32(const float* base, int ld,
                                                   int lane) {
  const int row = lane & 15;
  const int kb  = (lane < 16) ? 0 : 8;
  const float* p0 = base + row * ld + kb;
  float4 f0 = *reinterpret_cast<const float4*>(p0);
  float4 f1 = *reinterpret_cast<const float4*>(p0 + 4);
  float4 f2 = *reinterpret_cast<const float4*>(p0 + 16);
  float4 f3 = *reinterpret_cast<const float4*>(p0 + 20);
  v16bf a;
  a[0]  = (__bf16)f0.x; a[1]  = (__bf16)f0.y;
  a[2]  = (__bf16)f0.z; a[3]  = (__bf16)f0.w;
  a[4]  = (__bf16)f1.x; a[5]  = (__bf16)f1.y;
  a[6]  = (__bf16)f1.z; a[7]  = (__bf16)f1.w;
  a[8]  = (__bf16)f2.x; a[9]  = (__bf16)f2.y;
  a[10] = (__bf16)f2.z; a[11] = (__bf16)f2.w;
  a[12] = (__bf16)f3.x; a[13] = (__bf16)f3.y;
  a[14] = (__bf16)f3.z; a[15] = (__bf16)f3.w;
  return a;
}

// B-matrix 32x16 fragment where logical B[k][n] = Mat[n*ld + k]  (B = Mat^T,
// Mat row-major NxK).  Per lane: K={0..7,16..23} (lane<16) or {8..15,24..31}.
static __device__ __forceinline__ v16bf load_bT_bf16(const bf16* mat, int ld,
                                                     int lane) {
  const int n  = lane & 15;
  const int kb = (lane < 16) ? 0 : 8;
  const bf16* p = mat + n * ld + kb;
  v8bf lo = *reinterpret_cast<const v8bf*>(p);
  v8bf hi = *reinterpret_cast<const v8bf*>(p + 16);
  return concat8(lo, hi);
}

// ---------------------------------------------------------------------------
// Kernel 1: fused QKV projection.  Q = (x Wq)/sqrt(H) [BT,H], K = x Wk [BT,H],
// Vt = (x Wv)^T [H,BT], all bf16.  128 rows per block, 8 waves x 16 rows.
// W k-chunks staged TRANSPOSED in LDS so B-fragments are ds_load_b128 pairs.
// ---------------------------------------------------------------------------
__global__ __launch_bounds__(256) void qkv_proj_kernel(
    const float* __restrict__ x, const float* __restrict__ Wq,
    const float* __restrict__ Wk, const float* __restrict__ Wv,
    bf16* __restrict__ Q, bf16* __restrict__ K, bf16* __restrict__ Vt) {
  __shared__ bf16 ldsWt[HEAD * 32];  // W chunk transposed: [h][k], 8 KB

  const int tid  = threadIdx.x;
  const int lane = tid & 31;
  const int wave = tid >> 5;
  const int row0 = blockIdx.x * 128 + wave * 16;

  const float qscale = 0.08838834764831845f;  // 1/sqrt(128), folded into Q

  for (int m = 0; m < 3; ++m) {
    const float* Wm = (m == 0) ? Wq : ((m == 1) ? Wk : Wv);

    v8f acc[8];
#pragma unroll
    for (int nt = 0; nt < 8; ++nt)
#pragma unroll
      for (int e = 0; e < 8; ++e) acc[nt][e] = 0.0f;

    for (int kc = 0; kc < N_EMBED; kc += 32) {
      __syncthreads();
      for (int i = tid; i < 32 * HEAD; i += 256) {
        const int k = i >> 7;          // 0..31 (coalesced global read)
        const int h = i & (HEAD - 1);  // 0..127
        ldsWt[h * 32 + k] = (__bf16)Wm[(kc + k) * HEAD + h];
      }
      if (kc + 32 < N_EMBED)
        __builtin_prefetch(&Wm[(kc + 32) * HEAD + tid], 0, 0);
      __syncthreads();

      v16bf a = load_a_f32(x + (long)row0 * N_EMBED + kc, N_EMBED, lane);
#pragma unroll
      for (int nt = 0; nt < 8; ++nt) {
        // B[k][n] = W[kc+k][nt*16+n] = ldsWt[(nt*16+n)*32 + k]
        v16bf b = load_bT_bf16(ldsWt + (nt * 16) * 32, 32, lane);
        acc[nt] = wmma_bf16(a, b, acc[nt]);
      }
    }

    const int col = lane & 15;
    if (m == 0) {
#pragma unroll
      for (int nt = 0; nt < 8; ++nt)
#pragma unroll
        for (int r = 0; r < 8; ++r) {
          const int row = row0 + ((lane < 16) ? r : 8 + r);
          Q[(long)row * HEAD + nt * 16 + col] = (__bf16)(acc[nt][r] * qscale);
        }
    } else if (m == 1) {
#pragma unroll
      for (int nt = 0; nt < 8; ++nt)
#pragma unroll
        for (int r = 0; r < 8; ++r) {
          const int row = row0 + ((lane < 16) ? r : 8 + r);
          K[(long)row * HEAD + nt * 16 + col] = (__bf16)acc[nt][r];
        }
    } else {
#pragma unroll
      for (int nt = 0; nt < 8; ++nt)
#pragma unroll
        for (int r = 0; r < 8; ++r) {
          const int row = row0 + ((lane < 16) ? r : 8 + r);
          Vt[(long)(nt * 16 + col) * BT + row] = (__bf16)acc[nt][r];
        }
    }
  }
}

// ---------------------------------------------------------------------------
// Kernel 2: flash attention.  One (batch, 128-query tile) per block.
// Wave w owns query rows [qt*128 + w*16, +16).  Key tiles are 64 wide.
// Softmax stats in VGPRs (half-wave shuffle reductions); LDS only holds the
// bf16 P tile for the C-layout -> A-layout transpose.
// ---------------------------------------------------------------------------
__global__ __launch_bounds__(256) void flash_attn_kernel(
    const bf16* __restrict__ Q, const bf16* __restrict__ K,
    const bf16* __restrict__ Vt, float* __restrict__ out) {
  __shared__ bf16 sPb[8][16][64];  // per-wave 16x64 P tile, bf16 (16 KB)

  const int lane  = threadIdx.x & 31;
  const int wave  = threadIdx.x >> 5;
  const int batch = blockIdx.y;
  const int qt    = blockIdx.x;
  const int qrow0 = qt * 128 + wave * 16;   // within batch
  const int grow0 = batch * SEQ + qrow0;    // global row
  const int col   = lane & 15;
  const int half  = (lane < 16) ? 0 : 8;    // C-layout row offset

  // Q fragments (16 rows x 128 head dim as 4 chunks) stay in VGPRs.
  v16bf qf[4];
#pragma unroll
  for (int hc = 0; hc < 4; ++hc)
    qf[hc] = load_a_bf16(Q + (long)grow0 * HEAD + hc * 32, HEAD, lane);

  v8f o[8];
#pragma unroll
  for (int ht = 0; ht < 8; ++ht)
#pragma unroll
    for (int e = 0; e < 8; ++e) o[ht][e] = 0.0f;

  // Register-resident softmax state, replicated across each 16-lane half.
  float m_run[8], l_run[8];
#pragma unroll
  for (int r = 0; r < 8; ++r) { m_run[r] = -INFINITY; l_run[r] = 0.0f; }

  const int nkt = 2 * qt + 2;  // causal: keys up to (qt+1)*128
  for (int kt = 0; kt < nkt; ++kt) {
    __syncthreads();
    const int kb0 = kt * 64;
    const bf16* Kb = K + (long)(batch * SEQ + kb0) * HEAD;

    // ---- S = Q K^T (16 rows x 64 keys), causal mask in registers ----------
    v8f s[4];
#pragma unroll
    for (int ks = 0; ks < 4; ++ks) {
#pragma unroll
      for (int e = 0; e < 8; ++e) s[ks][e] = 0.0f;
#pragma unroll
      for (int hc = 0; hc < 4; ++hc) {
        v16bf bk = load_bT_bf16(Kb + (ks * 16) * HEAD + hc * 32, HEAD, lane);
        s[ks] = wmma_bf16(qf[hc], bk, s[ks]);
      }
      const int key = kb0 + ks * 16 + col;
#pragma unroll
      for (int r = 0; r < 8; ++r) {
        const int row = qrow0 + half + r;
        s[ks][r] = (key <= row) ? s[ks][r] : -INFINITY;
      }
    }

    // ---- online softmax: row max / alpha via half-wave butterflies --------
    float mx[8], alpha[8];
#pragma unroll
    for (int r = 0; r < 8; ++r) {
      float m0 = fmaxf(fmaxf(s[0][r], s[1][r]), fmaxf(s[2][r], s[3][r]));
      m0 = fmaxf(m0, __shfl_xor(m0, 1, 32));
      m0 = fmaxf(m0, __shfl_xor(m0, 2, 32));
      m0 = fmaxf(m0, __shfl_xor(m0, 4, 32));
      m0 = fmaxf(m0, __shfl_xor(m0, 8, 32));
      mx[r]    = fmaxf(m_run[r], m0);
      alpha[r] = __expf(m_run[r] - mx[r]);
      m_run[r] = mx[r];
    }

    // ---- exp, P -> LDS (bf16), row sums -----------------------------------
#pragma unroll
    for (int r = 0; r < 8; ++r) {
      const int rl = half + r;
      float ls = 0.0f;
#pragma unroll
      for (int ks = 0; ks < 4; ++ks) {
        const float p = __expf(s[ks][r] - mx[r]);
        ls += p;
        sPb[wave][rl][ks * 16 + col] = (__bf16)p;
      }
      ls += __shfl_xor(ls, 1, 32);
      ls += __shfl_xor(ls, 2, 32);
      ls += __shfl_xor(ls, 4, 32);
      ls += __shfl_xor(ls, 8, 32);
      l_run[r] = l_run[r] * alpha[r] + ls;
    }

    // ---- rescale O by alpha -----------------------------------------------
#pragma unroll
    for (int ht = 0; ht < 8; ++ht)
#pragma unroll
      for (int r = 0; r < 8; ++r) o[ht][r] *= alpha[r];

    __syncthreads();  // P tile visible to cross-lane A-fragment loads

    // ---- O += P V  (V stored transposed: Vt[h][key]) ----------------------
#pragma unroll
    for (int kc = 0; kc < 2; ++kc) {
      v16bf pa = load_a_bf16(&sPb[wave][0][0] + kc * 32, 64, lane);
#pragma unroll
      for (int ht = 0; ht < 8; ++ht) {
        // B[k][n] = V[kb0+kc*32+k][ht*16+n] = Vt[(ht*16+n)*BT + key]
        const bf16* Vtile =
            Vt + (long)(ht * 16) * BT + (batch * SEQ + kb0 + kc * 32);
        v16bf bv = load_bT_bf16(Vtile, BT, lane);
        o[ht] = wmma_bf16(pa, bv, o[ht]);
      }
    }
  }

  // ---- finalize: divide by l and store f32 --------------------------------
  float lr[8];
#pragma unroll
  for (int r = 0; r < 8; ++r) lr[r] = 1.0f / l_run[r];
#pragma unroll
  for (int ht = 0; ht < 8; ++ht)
#pragma unroll
    for (int r = 0; r < 8; ++r) {
      const int rl = half + r;
      out[(long)(grow0 + rl) * HEAD + ht * 16 + col] = o[ht][r] * lr[r];
    }
}

// ---------------------------------------------------------------------------
extern "C" void kernel_launch(void* const* d_in, const int* in_sizes, int n_in,
                              void* d_out, int out_size, void* d_ws,
                              size_t ws_size, hipStream_t stream) {
  (void)in_sizes; (void)n_in; (void)out_size; (void)ws_size;
  const float* x  = (const float*)d_in[0];
  const float* Wq = (const float*)d_in[1];
  const float* Wk = (const float*)d_in[2];
  const float* Wv = (const float*)d_in[3];
  float* out = (float*)d_out;

  // Workspace: Q [BT,H], K [BT,H], Vt [H,BT], bf16 (4 MB each).
  bf16* Qw  = (bf16*)d_ws;
  bf16* Kw  = Qw + (size_t)BT * HEAD;
  bf16* Vtw = Kw + (size_t)BT * HEAD;

  qkv_proj_kernel<<<dim3(BT / 128), dim3(256), 0, stream>>>(
      x, Wq, Wk, Wv, Qw, Kw, Vtw);
  flash_attn_kernel<<<dim3(SEQ / 128, NBATCH), dim3(256), 0, stream>>>(
      Qw, Kw, Vtw, out);
}
